// Glm4MoE_27582279975510
// MI455X (gfx1250) — matrix operand order repaired and checked by reading, not compiled, not run on hardware
//
#include <hip/hip_runtime.h>
#include <hip/hip_bf16.h>
#include <math.h>

// ---------------------------------------------------------------------------
// GLM4-MoE layer for MI455X (gfx1250, wave32, WMMA).
// T=8192 H=2048 I=1024 E=64 K=8 G=8 TG=4.
// bf16 WMMA (16x16x32) + fp32 accumulate keeps the kernel HBM-bound
// (~77us floor on 1.8GB fp32 weights @ 23.3TB/s).  fp32->bf16 conversion is
// done exactly once per tile element at LDS-staging time; the WMMA consume
// path is ds_load_b128 for A, strided 16-bit LDS loads for B -> v_wmma.  LDS tiles are double-buffered so the
// HBM weight stream overlaps the matrix math.
// ---------------------------------------------------------------------------

#define T_ 8192
#define H_ 2048
#define I_ 1024
#define E_ 64
#define K_ 8
#define G_ 8
#define TG_ 4
#define ROUTED_SCALE 2.5f

#define CHUNKS 64
#define TOK_PER_CHUNK (T_ / CHUNKS)            /* 128  */
#define ENT_PER_CHUNK (TOK_PER_CHUNK * K_)     /* 1024 */

typedef __attribute__((ext_vector_type(16))) __bf16 v16bf;
typedef __attribute__((ext_vector_type(8)))  __bf16 v8bf;   // 16B = 1x b128
typedef __attribute__((ext_vector_type(8)))  float  v8f;

#define AS_STRIDE 40     /* bf16 elems per A row  (20 dwords)  */
#define BS1_STRIDE 264   /* bf16 elems per B row, gemm1 (132 dwords) */
#define BS2_STRIDE 136   /* bf16 elems per B row, gemm2 (68 dwords)  */

__device__ __forceinline__ v16bf cat16(v8bf lo, v8bf hi) {
  return __builtin_shufflevector(lo, hi, 0, 1, 2, 3, 4, 5, 6, 7,
                                 8, 9, 10, 11, 12, 13, 14, 15);
}

__device__ __forceinline__ v8bf cvt8(float4 a, float4 b) {
  v8bf r;
  r[0] = (__bf16)a.x; r[1] = (__bf16)a.y; r[2] = (__bf16)a.z; r[3] = (__bf16)a.w;
  r[4] = (__bf16)b.x; r[5] = (__bf16)b.y; r[6] = (__bf16)b.z; r[7] = (__bf16)b.w;
  return r;
}

// ---------------------------------------------------------------------------
// Fragment loaders (CDNA5 ISA 7.12.2 layouts, wave32), LDS already bf16:
//  A (16x32): lanes 0-15 -> M=lane, elems 0..7 = K 0..7, 8..15 = K 16..23
//             lanes 16-31 -> M=lane-16, elems 0..7 = K 8..15, 8..15 = K 24..31
//  B (32x16): N = lane&15, K order same as A (per lane half)
//  C/D (16x16 f32): vgpr v: lanes0-15 (M=v,N=lane); lanes16-31 (M=v+8,N=lane-16)
// ---------------------------------------------------------------------------
__device__ __forceinline__ v16bf a_frag_ld(const __bf16* __restrict__ As,
                                           int lane) {
  const int m  = lane & 15;
  const int kh = (lane >> 4) * 8;
  const __bf16* r = As + m * AS_STRIDE;
  v8bf lo = *(const v8bf*)(r + kh);
  v8bf hi = *(const v8bf*)(r + 16 + kh);
  return cat16(lo, hi);
}

__device__ __forceinline__ v16bf b_frag_ld(const __bf16* __restrict__ Bs,
                                           int lane, int stride, int n0) {
  const int kh = (lane >> 4) * 8;
  const __bf16* r = Bs + n0 + (lane & 15);
  v16bf b;
#pragma unroll
  for (int i = 0; i < 8; ++i) {
    b[i]     = r[(kh + i) * stride];
    b[8 + i] = r[(16 + kh + i) * stride];
  }
  return b;
}

__device__ __forceinline__ v8f wmma_bf16(v16bf a, v16bf b, v8f c) {
  return __builtin_amdgcn_wmma_f32_16x16x32_bf16(false, a, false, b,
                                                 (short)0, c, false, false);
}

// ---------------------------------------------------------------------------
// 1) Router: logits, sigmoid scoring, grouped top-k.  64 threads per token.
// ---------------------------------------------------------------------------
__global__ __launch_bounds__(64)
void router_topk_kernel(const float* __restrict__ x,
                        const float* __restrict__ gate_w,
                        const float* __restrict__ e_bias,
                        int*   __restrict__ topk_ids,
                        float* __restrict__ topk_w) {
  __shared__ float xrow[H_];
  __shared__ float score[E_];
  __shared__ float biased[E_];
  __shared__ float gscore[G_];
  __shared__ int   gsel[G_];

  const int t = blockIdx.x;
  const int e = threadIdx.x;

#pragma unroll 4
  for (int j = 0; j < H_ / 64; ++j)
    xrow[j * 64 + e] = x[(size_t)t * H_ + j * 64 + e];
  __syncthreads();

  const float* wrow = gate_w + (size_t)e * H_;
  float dot = 0.f;
  for (int h = 0; h < H_; h += 4) {
    dot += xrow[h] * wrow[h] + xrow[h + 1] * wrow[h + 1] +
           xrow[h + 2] * wrow[h + 2] + xrow[h + 3] * wrow[h + 3];
  }
  const float s = 1.f / (1.f + __expf(-dot));
  score[e]  = s;
  biased[e] = s + e_bias[e];
  __syncthreads();

  if (e < G_) {  // group score = sum of top-2 biased scores in group
    float m1 = -INFINITY, m2 = -INFINITY;
#pragma unroll
    for (int j = 0; j < E_ / G_; ++j) {
      float v = biased[e * (E_ / G_) + j];
      if (v > m1) { m2 = m1; m1 = v; }
      else if (v > m2) { m2 = v; }
    }
    gscore[e] = m1 + m2;
    gsel[e]   = 0;
  }
  __syncthreads();

  if (e == 0) {
    for (int r = 0; r < TG_; ++r) {       // top-TG groups (ties: lowest idx)
      float best = -INFINITY; int bi = 0;
      for (int g = 0; g < G_; ++g)
        if (!gsel[g] && gscore[g] > best) { best = gscore[g]; bi = g; }
      gsel[bi] = 1;
    }
    unsigned long long taken = 0ull;
    int   sid[K_];
    float sw[K_];
    float wsum = 0.f;
    for (int r = 0; r < K_; ++r) {
      float best = -INFINITY; int bi = 0;
      for (int ee = 0; ee < E_; ++ee) {
        if (!gsel[ee / (E_ / G_)]) continue;
        if ((taken >> ee) & 1ull) continue;
        if (biased[ee] > best) { best = biased[ee]; bi = ee; }
      }
      taken |= (1ull << bi);
      sid[r] = bi;
      sw[r]  = score[bi];     // combine weights from UNBIASED scores
      wsum  += sw[r];
    }
    const float inv = 1.f / wsum;
    for (int r = 0; r < K_; ++r) {
      topk_ids[(size_t)t * K_ + r] = sid[r];
      topk_w[(size_t)t * K_ + r]   = sw[r] * inv;
    }
  }
}

// ---------------------------------------------------------------------------
// 2) Deterministic (token-ordered) expert gather lists.
// ---------------------------------------------------------------------------
__global__ __launch_bounds__(64)
void count_kernel(const int* __restrict__ topk_ids, int* __restrict__ cnt) {
  __shared__ int ids[ENT_PER_CHUNK];
  const int c = blockIdx.x, e = threadIdx.x;
  const int* src = topk_ids + (size_t)c * ENT_PER_CHUNK;
  for (int j = 0; j < ENT_PER_CHUNK / 64; ++j) ids[j * 64 + e] = src[j * 64 + e];
  __syncthreads();
  int n = 0;
  for (int r = 0; r < ENT_PER_CHUNK; ++r) n += (ids[r] == e);
  cnt[c * E_ + e] = n;
}

__global__ __launch_bounds__(64)
void scan_kernel(const int* __restrict__ cnt, int* __restrict__ expert_off,
                 int* __restrict__ chunk_base) {
  __shared__ int tot[E_];
  __shared__ int off[E_ + 1];
  const int e = threadIdx.x;
  int s = 0;
  for (int c = 0; c < CHUNKS; ++c) s += cnt[c * E_ + e];
  tot[e] = s;
  __syncthreads();
  if (e == 0) {
    off[0] = 0;
    for (int i = 0; i < E_; ++i) off[i + 1] = off[i] + tot[i];
    for (int i = 0; i <= E_; ++i) expert_off[i] = off[i];
  }
  __syncthreads();
  int base = off[e];
  for (int c = 0; c < CHUNKS; ++c) {
    chunk_base[c * E_ + e] = base;
    base += cnt[c * E_ + e];
  }
}

__global__ __launch_bounds__(64)
void fill_kernel(const int* __restrict__ topk_ids,
                 const float* __restrict__ topk_w,
                 const int* __restrict__ chunk_base,
                 int* __restrict__ row_tok, float* __restrict__ row_w) {
  __shared__ int   ids[ENT_PER_CHUNK];
  __shared__ float wv[ENT_PER_CHUNK];
  const int c = blockIdx.x, e = threadIdx.x;
  const int*   src  = topk_ids + (size_t)c * ENT_PER_CHUNK;
  const float* wsrc = topk_w  + (size_t)c * ENT_PER_CHUNK;
  for (int j = 0; j < ENT_PER_CHUNK / 64; ++j) {
    ids[j * 64 + e] = src[j * 64 + e];
    wv[j * 64 + e]  = wsrc[j * 64 + e];
  }
  __syncthreads();
  int pos = chunk_base[c * E_ + e];
  const int tok0 = c * TOK_PER_CHUNK;
  for (int r = 0; r < ENT_PER_CHUNK; ++r) {
    if (ids[r] == e) {
      row_tok[pos] = tok0 + (r >> 3);   // r/K_
      row_w[pos]   = wv[r];
      ++pos;
    }
  }
}

// ---------------------------------------------------------------------------
// 3) GEMM1 + SiLU*mul.  WG tile: 64 rows x 128 act-cols (g and u slices).
//    Double-buffered bf16 LDS tiles; convert fp32->bf16 once at staging.
// ---------------------------------------------------------------------------
template <bool GATHER>
__global__ __launch_bounds__(256)
void gemm1_silu_kernel(const float* __restrict__ X,
                       const float* __restrict__ Wbase,
                       float* __restrict__ act,
                       const int* __restrict__ expert_off,
                       const int* __restrict__ row_tok) {
  __shared__ __bf16 As[2][64][AS_STRIDE];      // 64x32 bf16 A tile (+pad)
  __shared__ __bf16 Bs[2][32][BS1_STRIDE];     // 32x256 bf16: [0,128)=g [128,256)=u
  __shared__ int    tokbuf[64];

  int rb, re;
  const float* W;
  const int* tokp = nullptr;
  if (GATHER) {
    const int e = blockIdx.z;
    rb = expert_off[e];
    re = expert_off[e + 1];
    W  = Wbase + (size_t)e * (size_t)H_ * (2 * I_);
    tokp = row_tok + rb;
  } else {
    rb = 0; re = T_;
    W  = Wbase;
  }
  const int rows = re - rb;
  const int row0 = blockIdx.y * 64;
  if (row0 >= rows) return;
  const int cb = blockIdx.x * 128;             // act col base

  const int tid = threadIdx.x;
  if (tid < 64) {
    int r = row0 + tid;
    if (r >= rows) r = rows - 1;
    tokbuf[tid] = GATHER ? tokp[r] : r;
  }
  __syncthreads();

  const int wave = tid >> 5, lane = tid & 31;
  const int rg = wave >> 1;                    // 0..3 row group (16 rows)
  const int cs = wave & 1;                     // 0..1 col slice (64 cols)

  v8f accg[4], accu[4];
#pragma unroll
  for (int j = 0; j < 4; ++j) { v8f z = {}; accg[j] = z; accu[j] = z; }

  const int arow = tid >> 2, acol = (tid & 3) * 8;   // A: 64 rows x 8 elems
  const size_t xbase = (size_t)tokbuf[arow] * H_;
  const int brow = tid >> 3, bc0 = (tid & 7) * 32;   // B: 32 rows x 32 elems
  const int wcol = (bc0 < 128) ? (cb + bc0) : (I_ + cb + (bc0 - 128));
  const float* const wrow = W + (size_t)brow * (2 * I_) + wcol;

  v8bf ra, rbv[4];
  // prologue: fetch + convert tile 0
  {
    const float* xr = X + xbase + acol;
    ra = cvt8(*(const float4*)(xr), *(const float4*)(xr + 4));
    const float* wr = wrow;
#pragma unroll
    for (int p = 0; p < 4; ++p)
      rbv[p] = cvt8(*(const float4*)(wr + p * 8), *(const float4*)(wr + p * 8 + 4));
  }
  *(v8bf*)&As[0][arow][acol] = ra;
#pragma unroll
  for (int p = 0; p < 4; ++p) *(v8bf*)&Bs[0][brow][bc0 + p * 8] = rbv[p];
  __syncthreads();

  const int NK = H_ / 32;
  for (int kt = 0; kt < NK; ++kt) {
    const int cur = kt & 1;
    // fetch + convert next tile while current computes
    if (kt + 1 < NK) {
      const int k0 = (kt + 1) * 32;
      const float* xr = X + xbase + k0 + acol;
      ra = cvt8(*(const float4*)(xr), *(const float4*)(xr + 4));
      const float* wr = wrow + (size_t)k0 * (2 * I_);
#pragma unroll
      for (int p = 0; p < 4; ++p)
        rbv[p] = cvt8(*(const float4*)(wr + p * 8), *(const float4*)(wr + p * 8 + 4));
      if (kt + 2 < NK)
        __builtin_prefetch(wr + (size_t)32 * (2 * I_), 0, 1);
    }

    const v16bf a = a_frag_ld(&As[cur][rg * 16][0], lane);
#pragma unroll
    for (int j = 0; j < 4; ++j) {
      v16bf bg = b_frag_ld(&Bs[cur][0][0], lane, BS1_STRIDE, cs * 64 + j * 16);
      accg[j] = wmma_bf16(a, bg, accg[j]);
    }
#pragma unroll
    for (int j = 0; j < 4; ++j) {
      v16bf bu = b_frag_ld(&Bs[cur][0][0], lane, BS1_STRIDE, 128 + cs * 64 + j * 16);
      accu[j] = wmma_bf16(a, bu, accu[j]);
    }

    if (kt + 1 < NK) {
      const int nxt = cur ^ 1;
      *(v8bf*)&As[nxt][arow][acol] = ra;
#pragma unroll
      for (int p = 0; p < 4; ++p) *(v8bf*)&Bs[nxt][brow][bc0 + p * 8] = rbv[p];
    }
    __syncthreads();
  }

  // epilogue: act = silu(g) * u
#pragma unroll
  for (int j = 0; j < 4; ++j) {
#pragma unroll
    for (int v = 0; v < 8; ++v) {
      const int m = v + ((lane >> 4) << 3);
      const int r = row0 + rg * 16 + m;
      if (r < rows) {
        const float g = accg[j][v];
        const float u = accu[j][v];
        const float a = g / (1.f + __expf(-g)) * u;
        const int n = cb + cs * 64 + j * 16 + (lane & 15);
        act[(size_t)(rb + r) * I_ + n] = a;
      }
    }
  }
}

// ---------------------------------------------------------------------------
// 4) GEMM2 (down proj).  WG tile: 64 rows x 128 H-cols, double-buffered.
//    GATHER: out[t] += ROUTED_SCALE * w * acc (one expert per launch ->
//    rows unique -> race-free, deterministic).  !GATHER: out[t] = acc (init).
// ---------------------------------------------------------------------------
template <bool GATHER>
__global__ __launch_bounds__(256)
void gemm2_kernel(const float* __restrict__ A,
                  const float* __restrict__ Wbase,
                  float* __restrict__ out,
                  const int* __restrict__ expert_off,
                  const int* __restrict__ row_tok,
                  const float* __restrict__ row_w,
                  int expert) {
  __shared__ __bf16 As[2][64][AS_STRIDE];
  __shared__ __bf16 Bs[2][32][BS2_STRIDE];     // 32x128 bf16 (+pad)
  __shared__ int    tokbuf[64];
  __shared__ float  wbuf[64];

  int rb, re;
  const float* W;
  if (GATHER) {
    rb = expert_off[expert];
    re = expert_off[expert + 1];
    W  = Wbase + (size_t)expert * I_ * H_;
  } else {
    rb = 0; re = T_;
    W  = Wbase;
  }
  const int rows = re - rb;
  const int row0 = blockIdx.y * 64;
  if (row0 >= rows) return;
  const int cb = blockIdx.x * 128;

  const int tid = threadIdx.x;
  if (tid < 64) {
    int r = row0 + tid;
    if (r >= rows) r = rows - 1;
    tokbuf[tid] = GATHER ? row_tok[rb + r] : r;
    wbuf[tid]   = GATHER ? (ROUTED_SCALE * row_w[rb + r]) : 1.f;
  }
  __syncthreads();

  const int wave = tid >> 5, lane = tid & 31;
  const int rg = wave >> 1, cs = wave & 1;

  v8f acc[4];
#pragma unroll
  for (int j = 0; j < 4; ++j) { v8f z = {}; acc[j] = z; }

  const int arow = tid >> 2, acol = (tid & 3) * 8;
  int ar = row0 + arow; if (ar >= rows) ar = rows - 1;
  const float* const arowp = A + (size_t)(rb + ar) * I_ + acol;
  const int brow = tid >> 3, bc0 = (tid & 7) * 16;
  const float* const wrow = W + (size_t)brow * H_ + cb + bc0;

  v8bf ra, rbv[2];
  {
    ra = cvt8(*(const float4*)(arowp), *(const float4*)(arowp + 4));
#pragma unroll
    for (int p = 0; p < 2; ++p)
      rbv[p] = cvt8(*(const float4*)(wrow + p * 8), *(const float4*)(wrow + p * 8 + 4));
  }
  *(v8bf*)&As[0][arow][acol] = ra;
#pragma unroll
  for (int p = 0; p < 2; ++p) *(v8bf*)&Bs[0][brow][bc0 + p * 8] = rbv[p];
  __syncthreads();

  const int NK = I_ / 32;
  for (int kt = 0; kt < NK; ++kt) {
    const int cur = kt & 1;
    if (kt + 1 < NK) {
      const int k0 = (kt + 1) * 32;
      const float* arp = arowp + k0;
      ra = cvt8(*(const float4*)(arp), *(const float4*)(arp + 4));
      const float* wr = wrow + (size_t)k0 * H_;
#pragma unroll
      for (int p = 0; p < 2; ++p)
        rbv[p] = cvt8(*(const float4*)(wr + p * 8), *(const float4*)(wr + p * 8 + 4));
      if (kt + 2 < NK)
        __builtin_prefetch(wr + (size_t)32 * H_, 0, 1);
    }

    const v16bf a = a_frag_ld(&As[cur][rg * 16][0], lane);
#pragma unroll
    for (int j = 0; j < 4; ++j) {
      v16bf b = b_frag_ld(&Bs[cur][0][0], lane, BS2_STRIDE, cs * 64 + j * 16);
      acc[j] = wmma_bf16(a, b, acc[j]);
    }

    if (kt + 1 < NK) {
      const int nxt = cur ^ 1;
      *(v8bf*)&As[nxt][arow][acol] = ra;
#pragma unroll
      for (int p = 0; p < 2; ++p) *(v8bf*)&Bs[nxt][brow][bc0 + p * 8] = rbv[p];
    }
    __syncthreads();
  }

#pragma unroll
  for (int j = 0; j < 4; ++j) {
#pragma unroll
    for (int v = 0; v < 8; ++v) {
      const int m = v + ((lane >> 4) << 3);
      const int r = row0 + rg * 16 + m;
      if (r < rows) {
        const int   t  = tokbuf[rg * 16 + m];
        const float sc = wbuf[rg * 16 + m];
        const int n = cb + cs * 64 + j * 16 + (lane & 15);
        const size_t oi = (size_t)t * H_ + n;
        const float val = sc * acc[j][v];
        out[oi] = GATHER ? (out[oi] + val) : val;
      }
    }
  }
}

// ---------------------------------------------------------------------------
// Launch.  Workspace (~303 MB): topk ids/w, histogram/scan, gather lists,
// gathered activations [T*K, I], shared intermediate [T, I].
// ---------------------------------------------------------------------------
extern "C" void kernel_launch(void* const* d_in, const int* in_sizes, int n_in,
                              void* d_out, int out_size, void* d_ws,
                              size_t ws_size, hipStream_t stream) {
  const float* x       = (const float*)d_in[0];
  const float* gate_w  = (const float*)d_in[1];
  const float* e_bias  = (const float*)d_in[2];
  const float* w_gu    = (const float*)d_in[3];
  const float* w_down  = (const float*)d_in[4];
  const float* sh_gu   = (const float*)d_in[5];
  const float* sh_down = (const float*)d_in[6];
  float* out = (float*)d_out;

  char* wp = (char*)d_ws;
  auto alloc = [&](size_t bytes) {
    char* p = wp;
    wp += (bytes + 255) & ~(size_t)255;
    return p;
  };
  int*   topk_ids   = (int*)  alloc((size_t)T_ * K_ * 4);
  float* topk_w     = (float*)alloc((size_t)T_ * K_ * 4);
  int*   cnt        = (int*)  alloc((size_t)CHUNKS * E_ * 4);
  int*   chunk_base = (int*)  alloc((size_t)CHUNKS * E_ * 4);
  int*   expert_off = (int*)  alloc((size_t)(E_ + 1) * 4);
  int*   row_tok    = (int*)  alloc((size_t)T_ * K_ * 4);
  float* row_w      = (float*)alloc((size_t)T_ * K_ * 4);
  float* act        = (float*)alloc((size_t)T_ * K_ * I_ * 4);
  float* tmp_sh     = (float*)alloc((size_t)T_ * I_ * 4);

  // 1) routing
  router_topk_kernel<<<T_, 64, 0, stream>>>(x, gate_w, e_bias, topk_ids, topk_w);
  // 2) deterministic gather lists
  count_kernel<<<CHUNKS, 64, 0, stream>>>(topk_ids, cnt);
  scan_kernel<<<1, 64, 0, stream>>>(cnt, expert_off, chunk_base);
  fill_kernel<<<CHUNKS, 64, 0, stream>>>(topk_ids, topk_w, chunk_base,
                                         row_tok, row_w);

  const dim3 blk(256);
  // 3) shared expert: gate_up+silu -> tmp, down -> out (initializes out)
  gemm1_silu_kernel<false><<<dim3(I_ / 128, T_ / 64, 1), blk, 0, stream>>>(
      x, sh_gu, tmp_sh, nullptr, nullptr);
  gemm2_kernel<false><<<dim3(H_ / 128, T_ / 64, 1), blk, 0, stream>>>(
      tmp_sh, sh_down, out, nullptr, nullptr, nullptr, 0);

  // 4) routed experts: all GEMM1s concurrent (disjoint act rows)
  gemm1_silu_kernel<true><<<dim3(I_ / 128, T_ / 64, E_), blk, 0, stream>>>(
      x, w_gu, act, expert_off, row_tok);
  // down-proj per expert, sequential on stream (rows unique within an expert
  // -> plain RMW accumulate, fully deterministic like the reference scan)
  for (int e = 0; e < E_; ++e)
    gemm2_kernel<true><<<dim3(H_ / 128, T_ / 64, 1), blk, 0, stream>>>(
        act, w_down, out, expert_off, row_tok, row_w, e);
}
